// HKVAttention_48868137893980
// MI455X (gfx1250) — matrix operand order repaired and checked by reading, not compiled, not run on hardware
//
#include <hip/hip_runtime.h>
#include <hip/hip_bf16.h>
#include <math.h>

// ---------------- types ----------------
typedef __bf16 bf16;
typedef __attribute__((ext_vector_type(16))) __bf16  v16bf;
typedef __attribute__((ext_vector_type(8)))  float   v8f;
typedef __attribute__((ext_vector_type(4)))  unsigned int uint4v;

#define EPS_   1e-5f
#define B_     2
#define S_     1024
#define P_     2
#define H_     1024
#define NKV_   8192
#define NH_    16
#define HD_    64
#define M4_    (B_*S_*P_)   /* 4096 rows of (B,S,P) */

// ---------------- fragment helpers ----------------
union Frag16 { v16bf v; uint4v q[2]; };

// A fragment, contiguous k order used by this file: lane(0-15)=row M, elems 0..15 = K[half*16 .. +15]
// (half = lane>=16)
__device__ __forceinline__ v16bf load_a_frag(const bf16* p, int half) {
  Frag16 f;
  f.q[0] = *reinterpret_cast<const uint4v*>(p + half * 16);
  f.q[1] = *reinterpret_cast<const uint4v*>(p + half * 16 + 8);
  return f.v;
}
// B fragment, same contiguous k order as A: lane&15 = col N, 16 contiguous K at offset half*16
__device__ __forceinline__ v16bf load_b_frag(const bf16* p) {
  Frag16 f;
  f.q[0] = *reinterpret_cast<const uint4v*>(p);
  f.q[1] = *reinterpret_cast<const uint4v*>(p + 8);
  return f.v;
}
__device__ __forceinline__ v8f v8f_zero() {
  v8f z;
#pragma unroll
  for (int i = 0; i < 8; ++i) z[i] = 0.f;
  return z;
}
__device__ __forceinline__ v8f wmma_bf16(v16bf a, v16bf b, v8f c) {
  return __builtin_amdgcn_wmma_f32_16x16x32_bf16(false, a, false, b, (short)0, c,
                                                 false, false);
}

// ---------------- small elementwise kernels ----------------
__global__ void f32_to_bf16_k(const float* __restrict__ in, bf16* __restrict__ out,
                              size_t n) {
  size_t i = (size_t)blockIdx.x * blockDim.x + threadIdx.x;
  if (i < n) out[i] = (bf16)in[i];
}

__global__ void rmsnorm_rows_k(const float* __restrict__ x, const float* __restrict__ w,
                               bf16* __restrict__ out, int cols) {
  int r = blockIdx.x;
  const float* xr = x + (size_t)r * cols;
  float ss = 0.f;
  for (int c = threadIdx.x; c < cols; c += blockDim.x) { float t = xr[c]; ss += t * t; }
#pragma unroll
  for (int m = 16; m; m >>= 1) ss += __shfl_xor(ss, m, 32);
  __shared__ float red[8];
  if ((threadIdx.x & 31) == 0) red[threadIdx.x >> 5] = ss;
  __syncthreads();
  float tot = 0.f;
#pragma unroll
  for (int i = 0; i < 8; ++i) tot += red[i];
  float inv = rsqrtf(tot / (float)cols + EPS_);
  for (int c = threadIdx.x; c < cols; c += blockDim.x)
    out[(size_t)r * cols + c] = (bf16)(xr[c] * inv * w[c]);
}

__global__ void build_kvq_k(const float* __restrict__ q, const bf16* __restrict__ kvn,
                            const int* __restrict__ choices, bf16* __restrict__ kvq) {
  size_t i = (size_t)blockIdx.x * blockDim.x + threadIdx.x;
  if (i >= (size_t)M4_ * H_) return;
  int m = (int)(i / H_), h = (int)(i % H_);
  kvq[i] = (bf16)(q[i] + (float)kvn[(size_t)choices[m] * H_ + h]);
}

__global__ void rope_inplace_k(float* __restrict__ x) {
  int idx = blockIdx.x * blockDim.x + threadIdx.x;  // M4_ * NH_ * 32
  if (idx >= M4_ * NH_ * 32) return;
  int d  = idx & 31;
  int nh = (idx >> 5) & (NH_ - 1);
  int m  = idx >> 9;
  int s  = (m / P_) % S_;
  float freq = __expf(-(float)d * (9.210340371976184f / 32.0f));  // 10000^(-d/32)
  float sn, cs;
  __sincosf((float)s * freq, &sn, &cs);
  size_t base = (size_t)m * H_ + nh * HD_ + d;
  float x1 = x[base], x2 = x[base + 32];
  x[base]      = x1 * cs - x2 * sn;
  x[base + 32] = x2 * cs + x1 * sn;
}

__global__ void transpose_v_k(const float* __restrict__ v, bf16* __restrict__ vt) {
  size_t i = (size_t)blockIdx.x * blockDim.x + threadIdx.x;
  const size_t total = (size_t)B_ * NH_ * P_ * HD_ * S_;
  if (i >= total) return;
  int s  = (int)(i % S_);
  int d  = (int)((i / S_) % HD_);
  int p  = (int)((i / ((size_t)S_ * HD_)) % P_);
  int nh = (int)((i / ((size_t)S_ * HD_ * P_)) % NH_);
  int b  = (int)(i / ((size_t)S_ * HD_ * P_ * NH_));
  vt[i] = (bf16)v[((size_t)((b * S_ + s) * P_ + p)) * H_ + nh * HD_ + d];
}

// ---------------- WMMA GEMM: C[MxN] = A[MxK] @ B[NxK]^T ----------------
// block = 128 threads (4 waves); each wave does a 16x64 strip; block tile 64x64.
// Software-pipelined: next k-step fragments are loaded before current WMMAs.
template <bool STORE_BF16>
__global__ void gemm_nt_k(const bf16* __restrict__ A, const bf16* __restrict__ Bm,
                          void* __restrict__ Cout, int M, int N, int K) {
  const int wave = threadIdx.x >> 5, lane = threadIdx.x & 31;
  const int half = lane >> 4, ln = lane & 15;
  const int m0 = blockIdx.y * 64 + wave * 16;
  const int n0 = blockIdx.x * 64;
  v8f acc[4];
#pragma unroll
  for (int g = 0; g < 4; ++g) acc[g] = v8f_zero();

  const bf16* arow = A + (size_t)(m0 + ln) * K;
  const bf16* brow[4];
#pragma unroll
  for (int g = 0; g < 4; ++g)
    brow[g] = Bm + (size_t)(n0 + g * 16 + ln) * K + half * 16;

  v16bf a = load_a_frag(arow, half);
  v16bf b[4];
#pragma unroll
  for (int g = 0; g < 4; ++g) b[g] = load_b_frag(brow[g]);

#pragma unroll 1
  for (int k = 32; k < K; k += 32) {
    v16bf an = load_a_frag(arow + k, half);
    v16bf bn[4];
#pragma unroll
    for (int g = 0; g < 4; ++g) bn[g] = load_b_frag(brow[g] + k);
    // pull the following k-step's cachelines toward L0 (global_prefetch_b8)
    __builtin_prefetch(arow + k + 32, 0, 3);
    __builtin_prefetch(brow[0] + k + 32, 0, 3);
#pragma unroll
    for (int g = 0; g < 4; ++g) acc[g] = wmma_bf16(a, b[g], acc[g]);
    a = an;
#pragma unroll
    for (int g = 0; g < 4; ++g) b[g] = bn[g];
  }
#pragma unroll
  for (int g = 0; g < 4; ++g) acc[g] = wmma_bf16(a, b[g], acc[g]);

#pragma unroll
  for (int g = 0; g < 4; ++g)
#pragma unroll
    for (int v = 0; v < 8; ++v) {
      size_t off = (size_t)(m0 + v + 8 * half) * N + n0 + g * 16 + ln;
      if (STORE_BF16) ((bf16*)Cout)[off] = (bf16)acc[g][v];
      else            ((float*)Cout)[off] = acc[g][v];
    }
}

// ---------------- fused scores (q @ K^T) + argmax over 8192 kv ----------------
// Orientation: A = KV tile (rows = kv idx), B = Q tile (cols = q rows), so the
// argmax over kv is a per-lane scan (no shuffles in the loop).
// One q-tile (16 rows) per BLOCK; the 4 waves split the 8192 kv entries
// (2048 each) and combine via LDS at the end (wave order = index order, so
// strict-> comparisons preserve the first-max tie-break of jnp.argmax).
// Inner loop is software-pipelined: A-frags of kv tile t+1 are issued before
// the WMMAs of tile t. K-chunk loop kept ROLLED to avoid LICM->scratch spills.
__global__ void scores_argmax_k(const bf16* __restrict__ Q, const bf16* __restrict__ Km,
                                int* __restrict__ choices) {
  const int wave = threadIdx.x >> 5, lane = threadIdx.x & 31;
  const int half = lane >> 4, ln = lane & 15;
  const int m0 = blockIdx.x * 16;

  float bestv = -__builtin_inff();
  int   bestn = 0;

  const bf16* qrow = Q + (size_t)(m0 + ln) * H_ + half * 16;  // B operand (col = q row)
  const int nt0 = wave * (NKV_ / 4), nt1 = nt0 + (NKV_ / 4);

#pragma unroll 1
  for (int ntb = nt0; ntb < nt1; ntb += 64) {  // 4 kv tiles of 16
    v8f acc[4];
#pragma unroll
    for (int t = 0; t < 4; ++t) acc[t] = v8f_zero();

#pragma unroll 1
    for (int kb = 0; kb < H_; kb += 128) {   // rolled: prevents LICM->scratch spill
      v16bf qb[4];
#pragma unroll
      for (int j = 0; j < 4; ++j) qb[j] = load_b_frag(qrow + kb + 32 * j);
      // pipelined over kv tiles: load tile t+1's A-frags before tile t's WMMAs
      v16bf af[4];
      {
        const bf16* krow = Km + (size_t)(ntb + ln) * H_ + kb;
#pragma unroll
        for (int j = 0; j < 4; ++j) af[j] = load_a_frag(krow + 32 * j, half);
      }
#pragma unroll
      for (int t = 0; t < 4; ++t) {
        v16bf afn[4];
        if (t < 3) {
          const bf16* krow = Km + (size_t)(ntb + (t + 1) * 16 + ln) * H_ + kb;
#pragma unroll
          for (int j = 0; j < 4; ++j) afn[j] = load_a_frag(krow + 32 * j, half);
        }
#pragma unroll
        for (int j = 0; j < 4; ++j) acc[t] = wmma_bf16(af[j], qb[j], acc[t]);
        if (t < 3) {
#pragma unroll
          for (int j = 0; j < 4; ++j) af[j] = afn[j];
        }
      }
    }
    // per-lane argmax update: lane's column = q row (m0+ln); rows = kv indices
#pragma unroll
    for (int t = 0; t < 4; ++t)
#pragma unroll
      for (int v = 0; v < 8; ++v) {
        int n = ntb + t * 16 + v + 8 * half;
        float val = acc[t][v];
        if (val > bestv) { bestv = val; bestn = n; }
      }
  }
  // combine the two halves (each half saw a different set of kv rows)
  {
    float ov = __shfl_xor(bestv, 16, 32);
    int   on = __shfl_xor(bestn, 16, 32);
    if (ov > bestv || (ov == bestv && on < bestn)) { bestv = ov; bestn = on; }
  }
  // combine the 4 waves via LDS (ascending wave = ascending kv index)
  __shared__ float sval[4][16];
  __shared__ int   sidx[4][16];
  if (lane < 16) { sval[wave][ln] = bestv; sidx[wave][ln] = bestn; }
  __syncthreads();
  if (threadIdx.x < 16) {
    float bv = sval[0][threadIdx.x];
    int   bn = sidx[0][threadIdx.x];
#pragma unroll
    for (int w = 1; w < 4; ++w) {
      float ov = sval[w][threadIdx.x];
      int   on = sidx[w][threadIdx.x];
      if (ov > bv || (ov == bv && on < bn)) { bv = ov; bn = on; }
    }
    choices[m0 + threadIdx.x] = bn;
  }
}

// ---------------- causal flash attention (per (b, head, p)) ----------------
// grid = (S/64, B*NH*P), 128 threads. Wave handles 16 queries, 32 keys / iter.
__global__ void flash_attn_k(const bf16* __restrict__ Q, const bf16* __restrict__ Kc,
                             const bf16* __restrict__ Vt, float* __restrict__ O) {
  __shared__ bf16 pshare[4 * 16 * 32];  // per-wave 16x32 probability tile
  const int wave = threadIdx.x >> 5, lane = threadIdx.x & 31;
  const int half = lane >> 4, ln = lane & 15;
  const int bnp = blockIdx.y;
  const int b  = bnp / (NH_ * P_);
  const int nh = (bnp / P_) % NH_;
  const int p  = bnp % P_;
  const int qs0 = blockIdx.x * 64 + wave * 16;
  bf16* myp = pshare + wave * 16 * 32;

  const size_t rowstride = (size_t)P_ * H_;  // step per seq position s
  const bf16* qbase = Q + ((size_t)(b * S_) * P_ + p) * H_ + nh * HD_;
  const bf16* kbase = Kc + ((size_t)(b * S_) * P_ + p) * H_ + nh * HD_;
  float*      obase = O + ((size_t)(b * S_) * P_ + p) * H_ + nh * HD_;
  const bf16* vtb   = Vt + (((size_t)(b * NH_ + nh) * P_ + p) * HD_) * S_;

  // query A-fragments (constant over the key loop), hd=64 -> two K=32 chunks
  const bf16* qr = qbase + (size_t)(qs0 + ln) * rowstride;
  v16bf qa0 = load_a_frag(qr + 0, half);
  v16bf qa1 = load_a_frag(qr + 32, half);

  v8f oacc[4];
#pragma unroll
  for (int g = 0; g < 4; ++g) oacc[g] = v8f_zero();
  float rmax[8], rsum[8];
#pragma unroll
  for (int v = 0; v < 8; ++v) { rmax[v] = -__builtin_inff(); rsum[v] = 0.f; }

  const int tend = qs0 + 15;
#pragma unroll 1
  for (int t0 = 0; t0 <= tend; t0 += 32) {
    // ---- S = Q K^T for 32 keys (two 16-col groups) ----
    v8f sa = v8f_zero(), sb = v8f_zero();
#pragma unroll
    for (int kk = 0; kk < 2; ++kk) {
      v16bf a = kk ? qa1 : qa0;
      const bf16* ka = kbase + (size_t)(t0 + ln)      * rowstride + kk * 32 + half * 16;
      const bf16* kb = kbase + (size_t)(t0 + 16 + ln) * rowstride + kk * 32 + half * 16;
      sa = wmma_bf16(a, load_b_frag(ka), sa);
      sb = wmma_bf16(a, load_b_frag(kb), sb);
    }
    // ---- causal mask + online softmax (row stats via shfl within 16-lane half) ----
#pragma unroll
    for (int v = 0; v < 8; ++v) {
      int sq = qs0 + v + 8 * half;
      float la = (t0 + ln      <= sq) ? sa[v] * 0.125f : -__builtin_inff();
      float lb = (t0 + 16 + ln <= sq) ? sb[v] * 0.125f : -__builtin_inff();
      float mx = fmaxf(la, lb);
#pragma unroll
      for (int m = 1; m < 16; m <<= 1) mx = fmaxf(mx, __shfl_xor(mx, m, 32));
      float nm = fmaxf(rmax[v], mx);
      float sc = __expf(rmax[v] - nm);  // first iter: exp(-inf)=0
      float pa = __expf(la - nm);
      float pb = __expf(lb - nm);
      float ps = pa + pb;
#pragma unroll
      for (int m = 1; m < 16; m <<= 1) ps += __shfl_xor(ps, m, 32);
      rsum[v] = rsum[v] * sc + ps;
      rmax[v] = nm;
#pragma unroll
      for (int g = 0; g < 4; ++g) oacc[g][v] *= sc;
      // D-layout -> LDS (row-major 16x32), to be re-read in A-layout
      myp[(v + 8 * half) * 32 + ln]      = (bf16)pa;
      myp[(v + 8 * half) * 32 + 16 + ln] = (bf16)pb;
    }
    asm volatile("s_wait_dscnt 0" ::: "memory");
    v16bf pf = load_a_frag(myp + (size_t)ln * 32, half);
    // ---- O += P @ V  (V pre-transposed: row d, contiguous t) ----
#pragma unroll
    for (int g = 0; g < 4; ++g) {
      const bf16* vp = vtb + (size_t)(g * 16 + ln) * S_ + t0 + half * 16;
      oacc[g] = wmma_bf16(pf, load_b_frag(vp), oacc[g]);
    }
  }
  // ---- normalize + store ----
#pragma unroll
  for (int g = 0; g < 4; ++g)
#pragma unroll
    for (int v = 0; v < 8; ++v) {
      int sq = qs0 + v + 8 * half;
      obase[(size_t)sq * rowstride + g * 16 + ln] = oacc[g][v] / rsum[v];
    }
}

// ---------------- host orchestration ----------------
extern "C" void kernel_launch(void* const* d_in, const int* in_sizes, int n_in,
                              void* d_out, int out_size, void* d_ws, size_t ws_size,
                              hipStream_t stream) {
  (void)in_sizes; (void)n_in; (void)out_size; (void)ws_size;
  const float* x_f = (const float*)d_in[0];
  const float* Wq  = (const float*)d_in[1];
  const float* Wk  = (const float*)d_in[2];
  const float* Wv  = (const float*)d_in[3];
  const float* Wo  = (const float*)d_in[4];
  const float* kv  = (const float*)d_in[5];
  const float* kvw = (const float*)d_in[6];
  float* out = (float*)d_out;

  char* ws = (char*)d_ws;
  const size_t MB = (size_t)1024 * 1024;
  bf16*  wq_bf   = (bf16*)(ws + 0 * MB);
  bf16*  wk_bf   = (bf16*)(ws + 2 * MB);
  bf16*  wv_bf   = (bf16*)(ws + 4 * MB);
  bf16*  wo_bf   = (bf16*)(ws + 6 * MB);
  bf16*  x_bf    = (bf16*)(ws + 8 * MB);    // 8MB;  later reused as kvq_bf / o_bf
  bf16*  kvn_bf  = (bf16*)(ws + 16 * MB);   // 16MB; later reused as o_f32
  float* q_f32   = (float*)(ws + 32 * MB);  // 16MB
  bf16*  q_bf    = (bf16*)(ws + 48 * MB);   // 8MB
  bf16*  kmat_bf = (bf16*)(ws + 56 * MB);   // 16MB; later reused as v_f32
  float* k_f32   = (float*)(ws + 72 * MB);  // 16MB
  bf16*  k_bf    = (bf16*)(ws + 88 * MB);   // 8MB
  bf16*  vt_bf   = (bf16*)(ws + 96 * MB);   // 8MB
  int*   choices = (int*)(ws + 104 * MB);   // 16KB

  bf16*  kvq_bf = x_bf;            // x dead after q GEMM
  float* v_f32  = (float*)kmat_bf; // kmat dead after argmax
  float* o_f32  = (float*)kvn_bf;  // kvn dead after kvq build
  bf16*  o_bf   = x_bf;            // kvq dead after k/v GEMMs

  const size_t nMH = (size_t)M4_ * H_;
  const size_t nHH = (size_t)H_ * H_;
  auto blocks = [](size_t n) { return dim3((unsigned)((n + 255) / 256)); };

  // 1) cast weights + inputs to bf16; rmsnorm kv
  f32_to_bf16_k<<<blocks(nHH), 256, 0, stream>>>(Wq, wq_bf, nHH);
  f32_to_bf16_k<<<blocks(nHH), 256, 0, stream>>>(Wk, wk_bf, nHH);
  f32_to_bf16_k<<<blocks(nHH), 256, 0, stream>>>(Wv, wv_bf, nHH);
  f32_to_bf16_k<<<blocks(nHH), 256, 0, stream>>>(Wo, wo_bf, nHH);
  f32_to_bf16_k<<<blocks(nMH), 256, 0, stream>>>(x_f, x_bf, nMH);
  rmsnorm_rows_k<<<NKV_, 256, 0, stream>>>(kv, kvw, kvn_bf, H_);

  // 2) q = x @ Wq^T
  gemm_nt_k<false><<<dim3(H_ / 64, M4_ / 64), 128, 0, stream>>>(x_bf, wq_bf, q_f32, M4_, H_, H_);
  f32_to_bf16_k<<<blocks(nMH), 256, 0, stream>>>(q_f32, q_bf, nMH);

  // 3) Kmat = msnorm(kv) @ Wk^T (bf16), then fused scores+argmax routing
  gemm_nt_k<true><<<dim3(H_ / 64, NKV_ / 64), 128, 0, stream>>>(kvn_bf, wk_bf, kmat_bf, NKV_, H_, H_);
  scores_argmax_k<<<M4_ / 16, 128, 0, stream>>>(q_bf, kmat_bf, choices);

  // 4) kvq = q + msnorm(kv[choices]);  k = kvq@Wk^T;  v = kvq@Wv^T
  build_kvq_k<<<blocks(nMH), 256, 0, stream>>>(q_f32, kvn_bf, choices, kvq_bf);
  gemm_nt_k<false><<<dim3(H_ / 64, M4_ / 64), 128, 0, stream>>>(kvq_bf, wk_bf, k_f32, M4_, H_, H_);
  gemm_nt_k<false><<<dim3(H_ / 64, M4_ / 64), 128, 0, stream>>>(kvq_bf, wv_bf, v_f32, M4_, H_, H_);

  // 5) RoPE on q,k; cast; transpose v to [b,h,p][d][s]
  rope_inplace_k<<<blocks((size_t)M4_ * NH_ * 32), 256, 0, stream>>>(q_f32);
  rope_inplace_k<<<blocks((size_t)M4_ * NH_ * 32), 256, 0, stream>>>(k_f32);
  f32_to_bf16_k<<<blocks(nMH), 256, 0, stream>>>(q_f32, q_bf, nMH);
  f32_to_bf16_k<<<blocks(nMH), 256, 0, stream>>>(k_f32, k_bf, nMH);
  transpose_v_k<<<blocks(nMH), 256, 0, stream>>>(v_f32, vt_bf);

  // 6) causal flash attention
  flash_attn_k<<<dim3(S_ / 64, B_ * NH_ * P_), 128, 0, stream>>>(q_bf, k_bf, vt_bf, o_f32);

  // 7) out = o @ Wo^T
  f32_to_bf16_k<<<blocks(nMH), 256, 0, stream>>>(o_f32, o_bf, nMH);
  gemm_nt_k<false><<<dim3(H_ / 64, M4_ / 64), 128, 0, stream>>>(o_bf, wo_bf, out, M4_, H_, H_);
}
